// TransformerClassification_75771813037019
// MI455X (gfx1250) — matrix operand order repaired and checked by reading, not compiled, not run on hardware
//
#include <hip/hip_runtime.h>
#include <hip/hip_bf16.h>

typedef __attribute__((ext_vector_type(16))) _Float16 v16h;
typedef __attribute__((ext_vector_type(8)))  _Float16 v8h;
typedef __attribute__((ext_vector_type(8)))  float    v8f;
typedef __attribute__((ext_vector_type(4)))  int      v4i;

#define EMBD 1024
#define SEQL 1024
#define NBAT 32
#define MROWS (NBAT * SEQL)   // 32768 folded (batch,seq) rows

// ---------------------------------------------------------------------------
// CDNA5 async global->LDS copy (ASYNCcnt path). The builtin on this toolchain
// takes (v4i addrspace(1)*, v4i addrspace(3)*, imm offset, imm cpol).
// Guarded so the file still compiles if the builtin is absent.
// ---------------------------------------------------------------------------
#if __has_builtin(__builtin_amdgcn_global_load_async_to_lds_b128)
#define HAVE_ASYNC_LDS 1
typedef __attribute__((address_space(1))) v4i g_v4i;
typedef __attribute__((address_space(3))) v4i l_v4i;
#else
#define HAVE_ASYNC_LDS 0
#endif

__device__ __forceinline__ void cp16_g2l(const _Float16* g, _Float16* l)
{
#if HAVE_ASYNC_LDS
    __builtin_amdgcn_global_load_async_to_lds_b128((g_v4i*)g, (l_v4i*)l, 0, 0);
#else
    *(v8h*)l = *(const v8h*)g;
#endif
}

__device__ __forceinline__ void async_drain()
{
#if HAVE_ASYNC_LDS
#if __has_builtin(__builtin_amdgcn_s_wait_asynccnt)
    __builtin_amdgcn_s_wait_asynccnt(0);
#else
    asm volatile("s_wait_asynccnt 0" ::: "memory");
#endif
#endif
}

// ---------------------------------------------------------------------------
// x1 = sqrt(E)*x + pe   (pe broadcast over batch)
// ---------------------------------------------------------------------------
__global__ void __launch_bounds__(256) embed_kernel(const float* __restrict__ x,
                                                    const float* __restrict__ pe,
                                                    float* __restrict__ out,
                                                    long long n)
{
    long long i = (long long)blockIdx.x * 256 + threadIdx.x;
    if (i < n) out[i] = 32.0f * x[i] + pe[i % (long long)(SEQL * EMBD)];
}

// f32 -> f16 (weights, one-time)
__global__ void __launch_bounds__(256) cvt_kernel(const float* __restrict__ in,
                                                  _Float16* __restrict__ out, int n)
{
    int i = blockIdx.x * 256 + threadIdx.x;
    if (i < n) out[i] = (_Float16)in[i];
}

// ---------------------------------------------------------------------------
// Row LayerNorm over E=1024, emits f16 for the WMMA GEMMs.
// ---------------------------------------------------------------------------
__global__ void __launch_bounds__(256) ln_kernel(const float* __restrict__ x,
                                                 const float* __restrict__ g,
                                                 const float* __restrict__ be,
                                                 _Float16* __restrict__ out)
{
    const long long row = blockIdx.x;
    const float* xr = x + row * EMBD;
    _Float16* orow  = out + row * EMBD;
    const int t = threadIdx.x;

    float lx[4];
    float s = 0.f, s2 = 0.f;
#pragma unroll
    for (int i = 0; i < 4; ++i) {
        float v = xr[t + 256 * i];
        lx[i] = v; s += v; s2 += v * v;
    }
    __shared__ float r1[256], r2[256];
    r1[t] = s; r2[t] = s2;
    __syncthreads();
    for (int off = 128; off > 0; off >>= 1) {
        if (t < off) { r1[t] += r1[t + off]; r2[t] += r2[t + off]; }
        __syncthreads();
    }
    const float mu  = r1[0] * (1.0f / EMBD);
    const float var = r2[0] * (1.0f / EMBD) - mu * mu;
    const float rs  = rsqrtf(var + 1e-5f);
#pragma unroll
    for (int i = 0; i < 4; ++i) {
        const int c = t + 256 * i;
        orow[c] = (_Float16)((lx[i] - mu) * rs * g[c] + be[c]);
    }
}

// ---------------------------------------------------------------------------
// Row softmax over S=1024 with scale 1/sqrt(E)=1/32. In-place on the d_out
// w-slice + f16 copy for the P.V GEMM.
// ---------------------------------------------------------------------------
__global__ void __launch_bounds__(256) softmax_kernel(float* __restrict__ wio,
                                                      _Float16* __restrict__ w16)
{
    const long long row = blockIdx.x;
    float* wr     = wio + row * SEQL;
    _Float16* hr  = w16 + row * SEQL;
    const int t = threadIdx.x;

    float lv[4];
    float mx = -3.4e38f;
#pragma unroll
    for (int i = 0; i < 4; ++i) {
        lv[i] = wr[t + 256 * i] * (1.0f / 32.0f);
        mx = fmaxf(mx, lv[i]);
    }
    __shared__ float red[256];
    red[t] = mx;
    __syncthreads();
    for (int off = 128; off > 0; off >>= 1) {
        if (t < off) red[t] = fmaxf(red[t], red[t + off]);
        __syncthreads();
    }
    mx = red[0];
    __syncthreads();

    float s = 0.f;
#pragma unroll
    for (int i = 0; i < 4; ++i) { lv[i] = __expf(lv[i] - mx); s += lv[i]; }
    red[t] = s;
    __syncthreads();
    for (int off = 128; off > 0; off >>= 1) {
        if (t < off) red[t] += red[t + off];
        __syncthreads();
    }
    const float inv = 1.0f / red[0];
#pragma unroll
    for (int i = 0; i < 4; ++i) {
        const float p = lv[i] * inv;
        wr[t + 256 * i] = p;
        hr[t + 256 * i] = (_Float16)p;
    }
}

// ---------------------------------------------------------------------------
// WMMA GEMM: C[M,N] = A[M,K](f16) * B + bias (+res) (relu) -> f32/f16.
//   BT=true : B stored [N,K] row-major (A @ W.T)
//   BT=false: B stored [K,N] row-major (A @ B)
// 256 threads = 8 wave32. Workgroup tile 128x128, K stepped by 32.
// Wave grid 4(M) x 2(N): each wave computes 32x64 = 2 A-frags x 4 B-frags
// -> 8 v_wmma_f32_16x16x32_f16 per K-step on 8 resident f32 accumulators.
// LDS is double-buffered (32KB); the next K-slice is fetched with
// GLOBAL_LOAD_ASYNC_TO_LDS_B128 (ASYNCcnt) while the current one feeds the
// WMMA pipe, drained with s_wait_asynccnt before the barrier.
// Fragment gathers use contiguous k = 16*hi + e for both A and B (contiguous 16B runs ->
// ds_load_b128). Uniform control flow -> EXEC all-ones around the WMMA ops.
// ---------------------------------------------------------------------------
template <bool BT>
__global__ void __launch_bounds__(256)
gemm_f16_wmma(const _Float16* __restrict__ A, const _Float16* __restrict__ Bm,
              const float* __restrict__ bias, const float* __restrict__ res32,
              float* __restrict__ out32, _Float16* __restrict__ out16, int relu,
              int M, int N, int K, long long sA, long long sB, long long sC)
{
    __shared__ alignas(16) _Float16 As[2][128][32];
    __shared__ alignas(16) _Float16 Bs[2][128][32];

    const int tid  = threadIdx.x;
    const int wave = tid >> 5;
    const int lane = tid & 31;
    const int hi   = lane >> 4;
    const int l15  = lane & 15;

    const int m0 = blockIdx.y * 128;
    const int n0 = blockIdx.x * 128;
    const long long zb = blockIdx.z;
    const _Float16* Az = A  + zb * sA + (long long)m0 * K;
    const _Float16* Bz = Bm + zb * sB;
    const long long cb = zb * sC;

    const int wm = (wave & 3) * 32;   // wave M offset in tile
    const int wn = (wave >> 2) * 64;  // wave N offset in tile

    v8f acc[8];
#pragma unroll
    for (int i = 0; i < 8; ++i) acc[i] = (v8f){};

    // staging coords: 128 rows x 32 halfs = 512 x 16B chunks, 2 per thread
    const int sr = tid >> 2;          // 0..63
    const int sc = (tid & 3) * 8;     // 0,8,16,24

    auto stage = [&](int b, int k0) {
        cp16_g2l(Az + (long long)sr * K + (k0 + sc),        &As[b][sr][sc]);
        cp16_g2l(Az + (long long)(sr + 64) * K + (k0 + sc), &As[b][sr + 64][sc]);
        if (BT) {
            const _Float16* bp = Bz + (long long)(n0 + sr) * K + (k0 + sc);
            cp16_g2l(bp,                     &Bs[b][sr][sc]);
            cp16_g2l(bp + (long long)64 * K, &Bs[b][sr + 64][sc]);
        } else {
            const int kk = tid & 31;
            const int nb = (tid >> 5) * 16;
            const _Float16* src = Bz + (long long)(k0 + kk) * N + (n0 + nb);
#pragma unroll
            for (int i = 0; i < 16; ++i) Bs[b][nb + i][kk] = src[i];
        }
    };

    auto compute = [&](int b) {
        v16h af[2], bf[4];
#pragma unroll
        for (int mi = 0; mi < 2; ++mi) {
            const int am = wm + 16 * mi + l15;
#pragma unroll
            for (int e = 0; e < 16; ++e) {
                const int ka = hi * 16 + e;
                af[mi][e] = As[b][am][ka];
            }
        }
#pragma unroll
        for (int ni = 0; ni < 4; ++ni) {
            const int bn = wn + 16 * ni + l15;
#pragma unroll
            for (int e = 0; e < 16; ++e) bf[ni][e] = Bs[b][bn][hi * 16 + e];
        }
#pragma unroll
        for (int mi = 0; mi < 2; ++mi)
#pragma unroll
            for (int ni = 0; ni < 4; ++ni)
                acc[mi * 4 + ni] = __builtin_amdgcn_wmma_f32_16x16x32_f16(
                    false, af[mi], false, bf[ni], (short)0, acc[mi * 4 + ni],
                    false, false);
    };

    // ---- double-buffered main loop with async global->LDS prefetch ----
    stage(0, 0);
    async_drain();
    __syncthreads();
    int buf = 0;
    for (int k0 = 0; k0 < K; k0 += 32) {
        const bool pre = (k0 + 32) < K;
        if (pre) stage(buf ^ 1, k0 + 32);   // async fetch next slice
        compute(buf);                       // 8 WMMAs on current slice
        if (pre) async_drain();             // s_wait_asynccnt 0
        __syncthreads();
        buf ^= 1;
    }

    // ---- epilogue: bias / residual(in-place-safe) / relu / f32 & f16 stores ----
#pragma unroll
    for (int mi = 0; mi < 2; ++mi) {
#pragma unroll
        for (int ni = 0; ni < 4; ++ni) {
            const int cn = n0 + wn + 16 * ni + l15;
            const v8f a = acc[mi * 4 + ni];
#pragma unroll
            for (int r = 0; r < 8; ++r) {
                const int cm = m0 + wm + 16 * mi + r + 8 * hi;
                float v = a[r];
                if (bias) v += bias[cn];
                const long long idx = cb + (long long)cm * N + cn;
                if (res32) v += res32[idx];
                if (relu)  v = fmaxf(v, 0.f);
                if (out32) out32[idx] = v;
                if (out16) out16[idx] = (_Float16)v;
            }
        }
    }
}

// ---------------------------------------------------------------------------
// logits[b,o] = x[b,0,:] . Wc[o,:] + bc[o]   (32x2 — trivial)
// ---------------------------------------------------------------------------
__global__ void __launch_bounds__(64) cls_kernel(const float* __restrict__ xa,
                                                 const float* __restrict__ Wc,
                                                 const float* __restrict__ bc,
                                                 float* __restrict__ logits)
{
    const int t = threadIdx.x;          // 0..63
    const int b = t >> 1, o = t & 1;
    const float* xr = xa + (long long)b * SEQL * EMBD;  // seq position 0
    const float* wr = Wc + o * EMBD;
    float s = bc[o];
    for (int e = 0; e < EMBD; ++e) s += xr[e] * wr[e];
    logits[t] = s;
}

// ---------------------------------------------------------------------------
extern "C" void kernel_launch(void* const* d_in, const int* in_sizes, int n_in,
                              void* d_out, int out_size, void* d_ws, size_t ws_size,
                              hipStream_t stream)
{
    (void)in_sizes; (void)n_in; (void)out_size; (void)ws_size;

    const float* x    = (const float*)d_in[0];
    const float* pe   = (const float*)d_in[1];
    const float* ln1g = (const float*)d_in[2];
    const float* ln1b = (const float*)d_in[3];
    const float* Wq   = (const float*)d_in[4];
    const float* bq   = (const float*)d_in[5];
    const float* Wk   = (const float*)d_in[6];
    const float* bk   = (const float*)d_in[7];
    const float* Wv   = (const float*)d_in[8];
    const float* bv   = (const float*)d_in[9];
    const float* Wo   = (const float*)d_in[10];
    const float* bo   = (const float*)d_in[11];
    const float* ln2g = (const float*)d_in[12];
    const float* ln2b = (const float*)d_in[13];
    const float* W1   = (const float*)d_in[14];
    const float* b1   = (const float*)d_in[15];
    const float* W2   = (const float*)d_in[16];
    const float* b2   = (const float*)d_in[17];
    const float* Wc   = (const float*)d_in[18];
    const float* bc   = (const float*)d_in[19];

    float* outF   = (float*)d_out;
    float* logits = outF;
    float* w1out  = outF + 64;
    float* w2out  = w1out + (long long)NBAT * SEQL * SEQL;

    const long long BSE = (long long)NBAT * SEQL * EMBD;  // 33.5M elems
    const long long BSS = (long long)NBAT * SEQL * SEQL;
    const int       WSZ = EMBD * EMBD;

    char* wp = (char*)d_ws;
    float*    xa   = (float*)wp;     wp += BSE * sizeof(float);   // residual stream (f32)
    _Float16* h16  = (_Float16*)wp;  wp += BSE * 2;               // LN output
    _Float16* q16  = (_Float16*)wp;  wp += BSE * 2;
    _Float16* k16  = (_Float16*)wp;  wp += BSE * 2;
    _Float16* v16  = (_Float16*)wp;  wp += BSE * 2;
    _Float16* t16  = (_Float16*)wp;  wp += BSE * 2;               // attn out / FF hidden
    _Float16* w16  = (_Float16*)wp;  wp += BSS * 2;               // softmax probs f16
    _Float16* Wq16 = (_Float16*)wp;  wp += (long long)WSZ * 2;
    _Float16* Wk16 = (_Float16*)wp;  wp += (long long)WSZ * 2;
    _Float16* Wv16 = (_Float16*)wp;  wp += (long long)WSZ * 2;
    _Float16* Wo16 = (_Float16*)wp;  wp += (long long)WSZ * 2;
    _Float16* W116 = (_Float16*)wp;  wp += (long long)WSZ * 2;
    _Float16* W216 = (_Float16*)wp;  wp += (long long)WSZ * 2;

    auto run_gemm = [&](bool bt, const _Float16* A, const _Float16* Bm,
                        const float* bias, const float* res, float* o32,
                        _Float16* o16, int relu, int M, int N, int K,
                        long long sA, long long sB, long long sC, int batches) {
        dim3 grid(N / 128, M / 128, batches), block(256);
        if (bt)
            gemm_f16_wmma<true><<<grid, block, 0, stream>>>(
                A, Bm, bias, res, o32, o16, relu, M, N, K, sA, sB, sC);
        else
            gemm_f16_wmma<false><<<grid, block, 0, stream>>>(
                A, Bm, bias, res, o32, o16, relu, M, N, K, sA, sB, sC);
    };

    // x1 = sqrt(E)*x + pe
    embed_kernel<<<(unsigned)((BSE + 255) / 256), 256, 0, stream>>>(x, pe, xa, BSE);

    // weights -> f16 (once)
    cvt_kernel<<<WSZ / 256, 256, 0, stream>>>(Wq, Wq16, WSZ);
    cvt_kernel<<<WSZ / 256, 256, 0, stream>>>(Wk, Wk16, WSZ);
    cvt_kernel<<<WSZ / 256, 256, 0, stream>>>(Wv, Wv16, WSZ);
    cvt_kernel<<<WSZ / 256, 256, 0, stream>>>(Wo, Wo16, WSZ);
    cvt_kernel<<<WSZ / 256, 256, 0, stream>>>(W1, W116, WSZ);
    cvt_kernel<<<WSZ / 256, 256, 0, stream>>>(W2, W216, WSZ);

    const long long sSE = (long long)SEQL * EMBD;
    const long long sSS = (long long)SEQL * SEQL;

    auto run_block = [&](float* wout) {
        // h = LN1(x) -> f16
        ln_kernel<<<MROWS, 256, 0, stream>>>(xa, ln1g, ln1b, h16);
        // q,k,v projections (A @ W.T + b), f16 outputs
        run_gemm(true, h16, Wq16, bq, nullptr, nullptr, q16, 0, MROWS, EMBD, EMBD, 0, 0, 0, 1);
        run_gemm(true, h16, Wk16, bk, nullptr, nullptr, k16, 0, MROWS, EMBD, EMBD, 0, 0, 0, 1);
        run_gemm(true, h16, Wv16, bv, nullptr, nullptr, v16, 0, MROWS, EMBD, EMBD, 0, 0, 0, 1);
        // raw scores = q @ k.T  (batched) straight into the d_out w-slice
        run_gemm(true, q16, k16, nullptr, nullptr, wout, nullptr, 0,
                 SEQL, SEQL, EMBD, sSE, sSE, sSS, NBAT);
        // softmax(scores/32) in place + f16 copy
        softmax_kernel<<<MROWS, 256, 0, stream>>>(wout, w16);
        // attn = P @ V (batched, B non-transposed)
        run_gemm(false, w16, v16, nullptr, nullptr, nullptr, t16, 0,
                 SEQL, EMBD, SEQL, sSS, sSE, sSE, NBAT);
        // x2 = x + attn @ Wo.T + bo   (in-place residual on xa)
        run_gemm(true, t16, Wo16, bo, xa, xa, nullptr, 0, MROWS, EMBD, EMBD, 0, 0, 0, 1);
        // h2 = LN2(x2)
        ln_kernel<<<MROWS, 256, 0, stream>>>(xa, ln2g, ln2b, h16);
        // ff hidden = relu(h2 @ W1.T + b1) -> f16
        run_gemm(true, h16, W116, b1, nullptr, nullptr, t16, 1, MROWS, EMBD, EMBD, 0, 0, 0, 1);
        // x3 = x2 + ff @ W2.T + b2      (in-place residual on xa)
        run_gemm(true, t16, W216, b2, xa, xa, nullptr, 0, MROWS, EMBD, EMBD, 0, 0, 0, 1);
    };

    run_block(w1out);   // block applied twice with same params
    run_block(w2out);

    cls_kernel<<<1, 64, 0, stream>>>(xa, Wc, bc, logits);
}
